// MoleculeMultiHeadSelfAttention_84026740179142
// MI455X (gfx1250) — matrix operand order repaired and checked by reading, not compiled, not run on hardware
//
#include <hip/hip_runtime.h>
#include <hip/hip_bf16.h>

// ---------------------------------------------------------------------------
// Fused molecule multi-head self-attention for MI455X (gfx1250, wave32).
// B=2048, N=50 (padded to 64), C=256, H=8, D=32.
// One workgroup per batch; wave w handles head w. All GEMMs via
// v_wmma_f32_16x16x32_bf16; all intermediates stay in 240KB LDS + VGPRs.
// ---------------------------------------------------------------------------

typedef __attribute__((ext_vector_type(16))) __bf16 v16bf;
typedef __attribute__((ext_vector_type(8)))  float  v8f;

#define NB   2048
#define NN   50
#define NC   256
#define NH   8
#define ND   32
#define NP   64      // N padded to multiple of 16 (and K=32 tiles)
#define WAPAD_ROWS 416  // H*N=400 rows padded to 416
#define WAPAD_COLS 64   // K=50 padded to 64

union V16U { v16bf v; unsigned int w[8]; unsigned short h[16]; };

static __device__ inline unsigned short f2bf_bits(float f) {
  union { __bf16 b; unsigned short u; } cv;
  cv.b = (__bf16)f;
  return cv.u;
}

// A-matrix (16x32 bf16), ISA layout: lane L: m=L&15, g=L>>4;
// VGPR v<4 : K = 8g + 2v {lo,hi};  VGPR v>=4 : K = 16 + 8g + 2(v-4).
// K-pairs are contiguous -> one b32 LDS read per VGPR.
static __device__ inline v16bf load_A16(const unsigned short* base, int row0,
                                        int stride, int k0, int lane) {
  const int m = lane & 15, g = lane >> 4;
  const unsigned short* row = base + (row0 + m) * stride + k0;
  V16U u;
#pragma unroll
  for (int v = 0; v < 4; ++v) {
    u.w[v]     = *(const unsigned int*)(row + g * 8 + 2 * v);
    u.w[v + 4] = *(const unsigned int*)(row + 16 + g * 8 + 2 * v);
  }
  return u.v;
}

// B-matrix (32x16 bf16): lane L: n=L&15, g=L>>4; VGPR v<4: K = 8g + 2v, v>=4: K = 16 + 8g + 2(v-4) {lo,hi}.
// rowptr already resolved for this lane's output column; K contiguous in mem.
static __device__ inline v16bf load_Brow(const unsigned short* rowptr, int k0, int g) {
  V16U u;
#pragma unroll
  for (int v = 0; v < 8; ++v)
    u.w[v] = *(const unsigned int*)(rowptr + k0 + (v >> 2) * 16 + g * 8 + 2 * (v & 3));
  return u.v;
}

// B-matrix where K strides through rows: element B[k,n] = base[k*stride + col].
static __device__ inline v16bf load_Bstrided(const unsigned short* base, int col,
                                             int stride, int k0, int g) {
  V16U u;
#pragma unroll
  for (int v = 0; v < 8; ++v) {
    int k = k0 + (v >> 2) * 16 + g * 8 + 2 * (v & 3);
    u.h[2 * v]     = base[k * stride + col];
    u.h[2 * v + 1] = base[(k + 1) * stride + col];
  }
  return u.v;
}

// A-matrix assembled from f32 LDS (mix buffer), converting to bf16 on the fly.
static __device__ inline v16bf load_Af32(const float* base, int row0,
                                         int stride, int k0, int lane) {
  const int m = lane & 15, g = lane >> 4;
  const float* row = base + (row0 + m) * stride + k0;
  V16U u;
#pragma unroll
  for (int v = 0; v < 4; ++v) {
    int kL = g * 8 + 2 * v, kH = 16 + g * 8 + 2 * v;
    u.h[2 * v]         = f2bf_bits(row[kL]);
    u.h[2 * v + 1]     = f2bf_bits(row[kL + 1]);
    u.h[8 + 2 * v]     = f2bf_bits(row[kH]);
    u.h[8 + 2 * v + 1] = f2bf_bits(row[kH + 1]);
  }
  return u.v;
}

static __device__ inline v8f wmma_bf16(v16bf a, v16bf b, v8f c) {
  return __builtin_amdgcn_wmma_f32_16x16x32_bf16(false, a, false, b,
                                                 (short)0, c, false, false);
}

// Row reductions across the 16 lanes of a half-wave (WMMA C-layout rows).
static __device__ inline float hmax16(float v) {
  v = fmaxf(v, __shfl_xor(v, 1));
  v = fmaxf(v, __shfl_xor(v, 2));
  v = fmaxf(v, __shfl_xor(v, 4));
  v = fmaxf(v, __shfl_xor(v, 8));
  return v;
}
static __device__ inline float hsum16(float v) {
  v += __shfl_xor(v, 1);
  v += __shfl_xor(v, 2);
  v += __shfl_xor(v, 4);
  v += __shfl_xor(v, 8);
  return v;
}

// ---------------------------------------------------------------------------
// Kernel 0: convert weights to bf16 into workspace (Wa/Wm zero-padded to
// [416][64] so the K=64 WMMA loop reads zeros past N=50).
// ---------------------------------------------------------------------------
__global__ void convert_weights(const float* __restrict__ Wq, const float* __restrict__ Wk,
                                const float* __restrict__ Wa, const float* __restrict__ Wm,
                                unsigned short* __restrict__ wq, unsigned short* __restrict__ wk,
                                unsigned short* __restrict__ wa, unsigned short* __restrict__ wm) {
  int i = blockIdx.x * blockDim.x + threadIdx.x;
  if (i < NC * NC) {
    wq[i] = f2bf_bits(Wq[i]);
    wk[i] = f2bf_bits(Wk[i]);
  }
  if (i < WAPAD_ROWS * WAPAD_COLS) {
    int r = i >> 6, c = i & 63;
    bool in = (r < NH * NN) & (c < NN);
    wa[i] = in ? f2bf_bits(Wa[r * NN + c]) : (unsigned short)0;
    wm[i] = in ? f2bf_bits(Wm[r * NN + c]) : (unsigned short)0;
  }
}

// ---------------------------------------------------------------------------
// Kernel 1: fully fused attention. 1 block = 1 batch, 256 threads = 8 waves.
// Dynamic LDS layout (bytes):
//   XB  @      0 : x[b]  bf16 [64][256]   (32 KB)
//   QB  @  32768 : q     bf16 [64][256]   (32 KB)  col = h*32 + d
//   KB  @  65536 : k(=v) bf16 [64][256]   (32 KB)
//   AJ  @  98304 : adj   bf16 [64][64]    ( 8 KB)
//   MT  @ 106496 : mat   bf16 [64][64]    ( 8 KB)
//   MIX @ 114688 : f32   [8 waves][64][64] (128 KB)
// total 245760 B (fits the 320 KB WGP LDS; 1 workgroup/WGP).
// ---------------------------------------------------------------------------
__global__ void __launch_bounds__(256, 1)
fused_mhsa(const float* __restrict__ x,   const float* __restrict__ adj,
           const float* __restrict__ mat,
           const float* __restrict__ bq,  const float* __restrict__ bk,
           const float* __restrict__ ba,  const float* __restrict__ bm,
           const float* __restrict__ lambdas,
           const unsigned short* __restrict__ Wq_bf, const unsigned short* __restrict__ Wk_bf,
           const unsigned short* __restrict__ Wa_bf, const unsigned short* __restrict__ Wm_bf,
           float* __restrict__ out) {
  extern __shared__ char smem[];
  unsigned short* XB = (unsigned short*)(smem);
  unsigned short* QB = (unsigned short*)(smem + 32768);
  unsigned short* KB = (unsigned short*)(smem + 65536);
  unsigned short* AJ = (unsigned short*)(smem + 98304);
  unsigned short* MT = (unsigned short*)(smem + 106496);
  float*         MIXALL = (float*)(smem + 114688);

  const int tid  = threadIdx.x;
  const int lane = tid & 31;
  const int w    = tid >> 5;       // wave id == head id
  const int g    = lane >> 4;      // half-wave
  const int ln   = lane & 15;
  const int b    = blockIdx.x;

  // ---- Phase 1: stage inputs as bf16 (zero-pad rows/cols >= 50) ----
  const float* xb = x + (size_t)b * (NN * NC);
  for (int i = tid; i < NP * NC; i += 256) {
    int r = i >> 8;
    XB[i] = (r < NN) ? f2bf_bits(xb[r * NC + (i & 255)]) : (unsigned short)0;
  }
  const float* ab = adj + (size_t)b * (NN * NN);
  const float* mb = mat + (size_t)b * (NN * NN);
  for (int i = tid; i < NP * NP; i += 256) {
    int r = i >> 6, c = i & 63;
    bool in = (r < NN) & (c < NN);
    AJ[i] = in ? f2bf_bits(ab[r * NN + c]) : (unsigned short)0;
    MT[i] = in ? f2bf_bits(mb[r * NN + c]) : (unsigned short)0;
  }
  __syncthreads();

  // ---- Phase 2: q/k projection (K=256). Wave w owns col-tiles {2w, 2w+1}. ----
#pragma unroll
  for (int mt = 0; mt < 4; ++mt) {
    v16bf A[8];
#pragma unroll
    for (int ks = 0; ks < 8; ++ks)
      A[ks] = load_A16(XB, mt * 16, NC, ks * 32, lane);
#pragma unroll
    for (int srcqk = 0; srcqk < 2; ++srcqk) {
      const unsigned short* W = srcqk ? Wk_bf : Wq_bf;
      const float* bias       = srcqk ? bk : bq;
      unsigned short* DST     = srcqk ? KB : QB;
#pragma unroll
      for (int nt = 0; nt < 2; ++nt) {
        const int col = (2 * w + nt) * 16 + ln;
        const unsigned short* wrow = W + col * NC;  // B[k=c, n] = W[col, c]
        v8f acc = {};
#pragma unroll
        for (int ks = 0; ks < 8; ++ks)
          acc = wmma_bf16(A[ks], load_Brow(wrow, ks * 32, g), acc);
        const float bcol = bias[col];
#pragma unroll
        for (int r = 0; r < 8; ++r) {
          int n = mt * 16 + r + 8 * g;
          DST[n * NC + col] = f2bf_bits(acc[r] + bcol);
        }
      }
    }
  }
  __syncthreads();

  // ---- Phase 3: per-head (wave w == head h) ----
  const int h = w;
  float* MIX = MIXALL + w * (NP * NP);

  // lambda softmax (uniform per wave -> scalar path)
  float a0 = lambdas[h * 3 + 0], a1 = lambdas[h * 3 + 1], a2 = lambdas[h * 3 + 2];
  float lm = fmaxf(a0, fmaxf(a1, a2));
  float e0 = __expf(a0 - lm), e1 = __expf(a1 - lm), e2 = __expf(a2 - lm);
  float li = 1.0f / (e0 + e1 + e2);
  const float l0 = e0 * li, l1 = e1 * li, l2 = e2 * li;

  const bool mcol = (ln >= 2);   // in col-tile 3, m = 48+ln >= 50 is padding
  v8f S[4][4];

  // 3a. attention scores: q[h] (16x32) x k[h]^T (32x16), K = D = 32
  {
    v16bf QA[4];
#pragma unroll
    for (int ti = 0; ti < 4; ++ti)
      QA[ti] = load_A16(QB, ti * 16, NC, h * ND, lane);
#pragma unroll
    for (int tj = 0; tj < 4; ++tj) {
      const unsigned short* krow = KB + (tj * 16 + ln) * NC;  // B[k=d, n=m]
      v16bf Bm = load_Brow(krow, h * ND, g);
#pragma unroll
      for (int ti = 0; ti < 4; ++ti) {
        v8f z = {};
        S[ti][tj] = wmma_bf16(QA[ti], Bm, z);
      }
    }
    const float scl = 0.0625f;   // C^-0.5 = 256^-0.5
#pragma unroll
    for (int ti = 0; ti < 4; ++ti) {
#pragma unroll
      for (int r = 0; r < 8; ++r) {
        float v0 = S[ti][0][r] * scl, v1 = S[ti][1][r] * scl;
        float v2 = S[ti][2][r] * scl;
        float v3 = mcol ? -3.0e38f : S[ti][3][r] * scl;
        float mx = hmax16(fmaxf(fmaxf(v0, v1), fmaxf(v2, v3)));
        v0 = __expf(v0 - mx); v1 = __expf(v1 - mx); v2 = __expf(v2 - mx);
        v3 = mcol ? 0.0f : __expf(v3 - mx);
        float inv = l0 / hsum16(v0 + v1 + v2 + v3);
        float* mrow = MIX + (ti * 16 + r + 8 * g) * NP + ln;
        mrow[0]  = inv * v0;  mrow[16] = inv * v1;
        mrow[32] = inv * v2;  mrow[48] = inv * v3;
      }
    }
  }

  // 3b. adj/mat head-projected softmaxes, accumulated into MIX
#pragma unroll
  for (int src = 0; src < 2; ++src) {
    const unsigned short* W = src ? Wm_bf : Wa_bf;
    const unsigned short* M = src ? MT : AJ;
    const float* bias       = src ? bm : ba;
    const float lc          = src ? l2 : l1;
    v16bf AA[4][2];
#pragma unroll
    for (int ti = 0; ti < 4; ++ti)
#pragma unroll
      for (int ks = 0; ks < 2; ++ks)
        AA[ti][ks] = load_A16(M, ti * 16, NP, ks * 32, lane);
    float bcol[4];
#pragma unroll
    for (int tj = 0; tj < 4; ++tj) {
      int m = tj * 16 + ln;
      bcol[tj] = (m < NN) ? bias[h * NN + m] : 0.0f;
      const unsigned short* wr = W + (h * NN + m) * WAPAD_COLS;  // B[k=j, n=m]
      v16bf B0 = load_Brow(wr, 0, g);
      v16bf B1 = load_Brow(wr, 32, g);
#pragma unroll
      for (int ti = 0; ti < 4; ++ti) {
        v8f acc = {};
        acc = wmma_bf16(AA[ti][0], B0, acc);
        acc = wmma_bf16(AA[ti][1], B1, acc);
        S[ti][tj] = acc;
      }
    }
#pragma unroll
    for (int ti = 0; ti < 4; ++ti) {
#pragma unroll
      for (int r = 0; r < 8; ++r) {
        float v0 = S[ti][0][r] + bcol[0], v1 = S[ti][1][r] + bcol[1];
        float v2 = S[ti][2][r] + bcol[2];
        float v3 = mcol ? -3.0e38f : S[ti][3][r] + bcol[3];
        float mx = hmax16(fmaxf(fmaxf(v0, v1), fmaxf(v2, v3)));
        v0 = __expf(v0 - mx); v1 = __expf(v1 - mx); v2 = __expf(v2 - mx);
        v3 = mcol ? 0.0f : __expf(v3 - mx);
        float q = lc / hsum16(v0 + v1 + v2 + v3);
        float* mrow = MIX + (ti * 16 + r + 8 * g) * NP + ln;
        mrow[0]  += q * v0;  mrow[16] += q * v1;
        mrow[32] += q * v2;  mrow[48] += q * v3;
      }
    }
  }

  // 3c. out = mix @ v (v = k), K = 64 (padded cols of mix are exact zeros)
#pragma unroll
  for (int dt = 0; dt < 2; ++dt) {
    const int dcol = h * ND + dt * 16 + ln;
    v16bf B0 = load_Bstrided(KB, dcol, NC, 0, g);   // B[k=m, n=d] = k[m, h*32+d]
    v16bf B1 = load_Bstrided(KB, dcol, NC, 32, g);
#pragma unroll
    for (int ti = 0; ti < 4; ++ti) {
      v16bf A0 = load_Af32(MIX, ti * 16, NP, 0, lane);
      v16bf A1 = load_Af32(MIX, ti * 16, NP, 32, lane);
      v8f acc = {};
      acc = wmma_bf16(A0, B0, acc);
      acc = wmma_bf16(A1, B1, acc);
#pragma unroll
      for (int r = 0; r < 8; ++r) {
        int n = ti * 16 + r + 8 * g;
        if (n < NN)
          out[(size_t)b * (NN * NC) + n * NC + dcol] = acc[r];
      }
    }
  }
}

// ---------------------------------------------------------------------------
extern "C" void kernel_launch(void* const* d_in, const int* in_sizes, int n_in,
                              void* d_out, int out_size, void* d_ws, size_t ws_size,
                              hipStream_t stream) {
  (void)in_sizes; (void)n_in; (void)out_size; (void)ws_size;
  const float* x       = (const float*)d_in[0];
  const float* adj     = (const float*)d_in[1];
  const float* matrix  = (const float*)d_in[2];
  const float* Wq      = (const float*)d_in[3];
  const float* bq      = (const float*)d_in[4];
  const float* Wk      = (const float*)d_in[5];
  const float* bk      = (const float*)d_in[6];
  const float* Wa      = (const float*)d_in[7];
  const float* ba      = (const float*)d_in[8];
  const float* Wm      = (const float*)d_in[9];
  const float* bm      = (const float*)d_in[10];
  const float* lambdas = (const float*)d_in[11];
  float* out = (float*)d_out;

  unsigned short* Wq_bf = (unsigned short*)d_ws;
  unsigned short* Wk_bf = Wq_bf + NC * NC;
  unsigned short* Wa_bf = Wk_bf + NC * NC;
  unsigned short* Wm_bf = Wa_bf + WAPAD_ROWS * WAPAD_COLS;

  const int smem_bytes = 245760;  // 240 KB dynamic LDS
  (void)hipFuncSetAttribute((const void*)fused_mhsa,
                            hipFuncAttributeMaxDynamicSharedMemorySize, smem_bytes);

  convert_weights<<<(NC * NC + 255) / 256, 256, 0, stream>>>(
      Wq, Wk, Wa, Wm, Wq_bf, Wk_bf, Wa_bf, Wm_bf);

  fused_mhsa<<<NB, 256, smem_bytes, stream>>>(
      x, adj, matrix, bq, bk, ba, bm, lambdas,
      Wq_bf, Wk_bf, Wa_bf, Wm_bf, out);
}
